// GAT_89043261980679
// MI455X (gfx1250) — matrix operand order repaired and checked by reading, not compiled, not run on hardware
//
#include <hip/hip_runtime.h>
#include <hip/hip_bf16.h>

typedef __attribute__((ext_vector_type(16))) _Float16 v16h;
typedef __attribute__((ext_vector_type(8)))  float    v8f;

// ---------------------------------------------------------------- fill
__global__ void gat_fill(float* __restrict__ p, float v, int n) {
  int i = blockIdx.x * blockDim.x + threadIdx.x;
  if (i < n) p[i] = v;
}

// ------------------------------------------------- pack W -> WMMA B tiles
// B operand layout (wave32, 16-bit, 32x16 KxN):
//   lane l, half h :  K = ks*32 + 16*(l>>4) + h ,  Ncol = tn*16 + (l&15)
// Stored as Bpk[((tn*4 + ks)*32 + lane)*16 + h], f16, zero-padded past N.
__global__ void gat_pack_b(const float* __restrict__ W, _Float16* __restrict__ Bpk,
                           int N, int tilesN) {
  int t = blockIdx.x * blockDim.x + threadIdx.x;          // (tn*4+ks)*32+lane
  int total = tilesN * 4 * 32;
  if (t >= total) return;
  int lane = t & 31;
  int ks   = (t >> 5) & 3;
  int tn   = (t >> 5) >> 2;
  int ncol = tn * 16 + (lane & 15);
  int k0   = ks * 32 + ((lane >> 4) << 4);
  _Float16* out = Bpk + (size_t)t * 16;
#pragma unroll
  for (int h = 0; h < 16; ++h) {
    float v = (ncol < N) ? W[(size_t)(k0 + h) * N + ncol] : 0.0f;
    out[h] = (_Float16)v;
  }
}

// ------------------------------------------------------------- WMMA GEMM
// D[M x N] = A[M x 128] * W[128 x N].  8 waves / block; one wave computes a
// full 16 x N output stripe: A tile converted to f16 ONCE (4 x v16h regs),
// packed B staged in LDS once per block.  TILES_N templated so the tn loop
// fully unrolls (32 / 12 straight-line v_wmma) and ds_load latency hides
// under the matrix pipe.  Requires M % 16 == 0 within active waves.
#define GEMM_WAVES 8
template <int TILES_N, bool FULL_N>
__global__ void __launch_bounds__(256)
gat_wmma_gemm(const float* __restrict__ A, const _Float16* __restrict__ Bpk,
              float* __restrict__ D, int M, int N) {
  __shared__ _Float16 Bs[TILES_N * 4 * 32 * 16];    // 32 KB (8) / 12 KB (3)

  int tid = threadIdx.x;
  constexpr int NVEC = (TILES_N * 4 * 32 * 16) / 8; // float4 chunks
  for (int i = tid; i < NVEC; i += 256)
    ((float4*)Bs)[i] = ((const float4*)Bpk)[i];
  __syncthreads();

  int wave  = tid >> 5;
  int lane  = tid & 31;
  int tm    = blockIdx.x * GEMM_WAVES + wave;
  if (tm * 16 >= M) return;                         // wave-uniform tail guard
  int lhalf = lane & 15;
  int lhi   = lane >> 4;

  const float* Arow = A + (size_t)(tm * 16 + lhalf) * 128;

  // A operand layout: lanes 0-15 hold row M=lane, K = {0..15} (+ks*32)
  //                   lanes 16-31 hold row M=lane-16, K = {16..31}
  v16h av[4];
#pragma unroll
  for (int ks = 0; ks < 4; ++ks) {
    int ka0 = ks * 32 + 16 * lhi;
    float af[16] __attribute__((aligned(16)));
    *(float4*)(af + 0)  = *(const float4*)(Arow + ka0 + 0);
    *(float4*)(af + 4)  = *(const float4*)(Arow + ka0 + 4);
    *(float4*)(af + 8)  = *(const float4*)(Arow + ka0 + 8);
    *(float4*)(af + 12) = *(const float4*)(Arow + ka0 + 12);
#pragma unroll
    for (int h = 0; h < 16; ++h) av[ks][h] = (_Float16)af[h];
  }

#pragma unroll
  for (int tn = 0; tn < TILES_N; ++tn) {
    const _Float16* bbase = Bs + (size_t)tn * 4 * 32 * 16 + (size_t)lane * 16;
    v8f c = {};
#pragma unroll
    for (int ks = 0; ks < 4; ++ks) {
      v16h bv = *(const v16h*)(bbase + ks * 32 * 16);
      c = __builtin_amdgcn_wmma_f32_16x16x32_f16(false, av[ks], false, bv,
                                                 (short)0, c, false, false);
    }
    // D layout: lane -> N = tn*16+(lane&15); vgpr r -> M = tm*16+8*(lane>>4)+r
    int n = tn * 16 + lhalf;
    float* drow = D + (size_t)(tm * 16 + 8 * lhi) * N + n;
    if (FULL_N || n < N) {
#pragma unroll
      for (int r = 0; r < 8; ++r)
        drow[(size_t)r * N] = c[r];
    }
  }
}

// ------------------------------------------------------- per-node alphas
__global__ void gat_alpha(const float* __restrict__ h, const float* __restrict__ aS,
                          const float* __restrict__ aD, float* __restrict__ asrc,
                          float* __restrict__ adst, int nNodes, int H, int C) {
  int i = blockIdx.x * blockDim.x + threadIdx.x;
  if (i >= nNodes * H) return;
  int node = i / H, hh = i - node * H;
  const float* hp  = h + (size_t)node * H * C + hh * C;
  const float* asp = aS + hh * C;
  const float* adp = aD + hh * C;
  float ss = 0.f, sd = 0.f;
  for (int c = 0; c < C; ++c) { float v = hp[c]; ss += v * asp[c]; sd += v * adp[c]; }
  asrc[i] = ss;
  adst[i] = sd;
}

// float atomic max via int-ordering trick
__device__ __forceinline__ void atomicMaxF32(float* addr, float val) {
  if (val >= 0.f) atomicMax((int*)addr, __float_as_int(val));
  else            atomicMin((unsigned int*)addr, (unsigned int)__float_as_int(val));
}

// --------------------------------------------- edge logits + segment max
__global__ void gat_edge_logits(const int* __restrict__ src, const int* __restrict__ dst,
                                const float* __restrict__ asrc, const float* __restrict__ adst,
                                float* __restrict__ e, float* __restrict__ m,
                                int E, int nNodes, int H) {
  int idx = blockIdx.x * blockDim.x + threadIdx.x;
  int Etot = E + nNodes;
  if (idx >= Etot * H) return;
  int ed = idx / H, hh = idx - ed * H;
  int s_, d_;
  if (ed < E) { s_ = src[ed]; d_ = dst[ed]; } else { s_ = d_ = ed - E; }
  float v = asrc[s_ * H + hh] + adst[d_ * H + hh];
  v = (v > 0.f) ? v : 0.2f * v;                 // leaky_relu(0.2)
  e[idx] = v;
  atomicMaxF32(&m[d_ * H + hh], v);
}

// ---------------------------------------------------- exp + segment sum
__global__ void gat_edge_expsum(const int* __restrict__ src, const int* __restrict__ dst,
                                const float* __restrict__ e, const float* __restrict__ m,
                                float* __restrict__ s, int E, int nNodes, int H) {
  int idx = blockIdx.x * blockDim.x + threadIdx.x;
  int Etot = E + nNodes;
  if (idx >= Etot * H) return;
  int ed = idx / H, hh = idx - ed * H;
  int d_ = (ed < E) ? dst[ed] : ed - E;
  float ex = expf(e[idx] - m[d_ * H + hh]);
  atomicAdd(&s[d_ * H + hh], ex);
}

// -------------------------------------- wave-per-edge message aggregation
// Lane handles 4 consecutive channels -> one global_load_b128 gather +
// 4 L2-resident GLOBAL_ATOMIC_ADD_F32 per lane. C is a multiple of 4, so a
// float4 never straddles a head boundary.
__global__ void gat_aggregate(const int* __restrict__ src, const int* __restrict__ dst,
                              const float* __restrict__ e, const float* __restrict__ m,
                              const float* __restrict__ s, const float* __restrict__ h,
                              float* __restrict__ agg, int E, int nNodes, int H, int C) {
  int wave = (int)(((size_t)blockIdx.x * blockDim.x + threadIdx.x) >> 5);
  int lane = threadIdx.x & 31;
  int Etot = E + nNodes;
  if (wave >= Etot) return;
  int s_, d_;
  if (wave < E) { s_ = src[wave]; d_ = dst[wave]; } else { s_ = d_ = wave - E; }
  float w0 = expf(e[wave * H + 0] - m[d_ * H + 0]) / (s[d_ * H + 0] + 1e-16f);
  float w1 = (H > 1) ? expf(e[wave * H + 1] - m[d_ * H + 1]) / (s[d_ * H + 1] + 1e-16f) : w0;
  int F = H * C;
  const float* hs = h + (size_t)s_ * F;
  float* ag = agg + (size_t)d_ * F;
  for (int c0 = lane * 4; c0 < F; c0 += 128) {
    float4 hv = *(const float4*)(hs + c0);
    float wv = (c0 < C) ? w0 : w1;
    atomicAdd(&ag[c0 + 0], hv.x * wv);
    atomicAdd(&ag[c0 + 1], hv.y * wv);
    atomicAdd(&ag[c0 + 2], hv.z * wv);
    atomicAdd(&ag[c0 + 3], hv.w * wv);
  }
}

// ------------------------------------------------------- bias (+ELU)
__global__ void gat_finalize(const float* __restrict__ agg, const float* __restrict__ b,
                             float* __restrict__ out, int nNodes, int F, int doElu) {
  int i = blockIdx.x * blockDim.x + threadIdx.x;
  if (i >= nNodes * F) return;
  float v = agg[i] + b[i % F];
  if (doElu) v = (v > 0.f) ? v : (expf(v) - 1.f);
  out[i] = v;
}

// =====================================================================
extern "C" void kernel_launch(void* const* d_in, const int* in_sizes, int n_in,
                              void* d_out, int out_size, void* d_ws, size_t ws_size,
                              hipStream_t stream) {
  (void)n_in; (void)out_size; (void)ws_size;
  const float* x   = (const float*)d_in[0];
  const int*   ei  = (const int*)d_in[1];
  const float* w1  = (const float*)d_in[2];
  const float* as1 = (const float*)d_in[3];
  const float* ad1 = (const float*)d_in[4];
  const float* b1  = (const float*)d_in[5];
  const float* w2  = (const float*)d_in[6];
  const float* as2 = (const float*)d_in[7];
  const float* ad2 = (const float*)d_in[8];
  const float* b2  = (const float*)d_in[9];
  const float* w3  = (const float*)d_in[10];
  const float* as3 = (const float*)d_in[11];
  const float* ad3 = (const float*)d_in[12];
  const float* b3  = (const float*)d_in[13];

  const int nNodes = in_sizes[0] / 128;
  const int E      = in_sizes[1] / 2;
  const int Etot   = E + nNodes;
  const int* srcIdx = ei;        // edge_index row 0
  const int* dstIdx = ei + E;    // edge_index row 1

  // ---- workspace carve (~85 MB) ----
  char* wsp = (char*)d_ws;
  auto carve = [&](size_t bytes) -> void* {
    void* p = (void*)wsp;
    wsp += (bytes + 255) & ~(size_t)255;
    return p;
  };
  float*    hbuf  = (float*)carve((size_t)nNodes * 128 * 4);
  float*    aggA  = (float*)carve((size_t)nNodes * 128 * 4);
  float*    aggB  = (float*)carve((size_t)nNodes * 128 * 4);
  float*    asrcb = (float*)carve((size_t)nNodes * 2 * 4);
  float*    adstb = (float*)carve((size_t)nNodes * 2 * 4);
  float*    mbuf  = (float*)carve((size_t)nNodes * 2 * 4);
  float*    sbuf  = (float*)carve((size_t)nNodes * 2 * 4);
  float*    ebuf  = (float*)carve((size_t)Etot * 2 * 4);
  _Float16* bpk   = (_Float16*)carve((size_t)8 * 4 * 32 * 16 * 2);

  const int tilesM = (nNodes + 15) / 16;   // 3125 (exact for 50000)

  auto layer = [&](const float* in, const float* W, const float* aS, const float* aD,
                   const float* bias, int H, int C, float* hout, float* agg,
                   float* actout, int doElu) {
    const int F = H * C;
    const int tilesN = (F + 15) / 16;
    { int t = tilesN * 4 * 32;
      gat_pack_b<<<(t + 255) / 256, 256, 0, stream>>>(W, bpk, F, tilesN); }
    { int blocks = (tilesM + GEMM_WAVES - 1) / GEMM_WAVES;
      if (tilesN == 8)
        gat_wmma_gemm<8, true><<<blocks, 256, 0, stream>>>(in, bpk, hout, nNodes, F);
      else
        gat_wmma_gemm<3, false><<<blocks, 256, 0, stream>>>(in, bpk, hout, nNodes, F); }
    { int t = nNodes * H;
      gat_alpha<<<(t + 255) / 256, 256, 0, stream>>>(hout, aS, aD, asrcb, adstb, nNodes, H, C);
      gat_fill<<<(t + 255) / 256, 256, 0, stream>>>(mbuf, -3.0e38f, t);
      gat_fill<<<(t + 255) / 256, 256, 0, stream>>>(sbuf, 0.0f, t); }
    { int t = nNodes * F;
      gat_fill<<<(t + 255) / 256, 256, 0, stream>>>(agg, 0.0f, t); }
    { int t = Etot * H;
      gat_edge_logits<<<(t + 255) / 256, 256, 0, stream>>>(srcIdx, dstIdx, asrcb, adstb,
                                                           ebuf, mbuf, E, nNodes, H);
      gat_edge_expsum<<<(t + 255) / 256, 256, 0, stream>>>(srcIdx, dstIdx, ebuf, mbuf,
                                                           sbuf, E, nNodes, H); }
    { int blocks = (int)(((size_t)Etot * 32 + 255) / 256);   // one wave per edge
      gat_aggregate<<<blocks, 256, 0, stream>>>(srcIdx, dstIdx, ebuf, mbuf, sbuf,
                                                hout, agg, E, nNodes, H, C); }
    { int t = nNodes * F;
      gat_finalize<<<(t + 255) / 256, 256, 0, stream>>>(agg, bias, actout, nNodes, F, doElu); }
  };

  layer(x,    w1, as1, ad1, b1, 2, 64, hbuf, aggA, aggA, 1);
  layer(aggA, w2, as2, ad2, b2, 2, 64, hbuf, aggB, aggB, 1);
  layer(aggB, w3, as3, ad3, b3, 1, 40, hbuf, aggA, (float*)d_out, 0);
}
